// GAT_29231547416617
// MI455X (gfx1250) — matrix operand order repaired and checked by reading, not compiled, not run on hardware
//
#include <hip/hip_runtime.h>
#include <hip/hip_bf16.h>

typedef __attribute__((ext_vector_type(16))) _Float16 v16h;
typedef __attribute__((ext_vector_type(8)))  _Float16 v8h;
typedef __attribute__((ext_vector_type(8)))  float    v8f;

#define IN_F   128
#define HID    16
#define NHEAD  8
#define OUT_F  40
#define NEG_SL 0.2f

// ---------- helpers: order-preserving float<->uint key for atomic max ----------
__device__ __forceinline__ unsigned fkey(float f) {
  unsigned b = __float_as_uint(f);
  return (b & 0x80000000u) ? ~b : (b | 0x80000000u);
}
__device__ __forceinline__ float fdec(unsigned k) {
  return (k & 0x80000000u) ? __uint_as_float(k & 0x7FFFFFFFu)
                           : __uint_as_float(~k);
}
__device__ __forceinline__ float lrelu(float v) { return v > 0.0f ? v : NEG_SL * v; }

// ---------- generic utility kernels ----------
__global__ void k_zero_u32(unsigned* __restrict__ p, int n) {
  int i = blockIdx.x * blockDim.x + threadIdx.x;
  if (i < n) p[i] = 0u;
}

__global__ void k_f32_to_f16(const float* __restrict__ in, _Float16* __restrict__ out, int n) {
  int i = blockIdx.x * blockDim.x + threadIdx.x;
  if (i < n) out[i] = (_Float16)in[i];
}

// convert f32 W[K,Ncols] (row-major) -> f16 BT[Ncols,K] (i.e. transposed)
__global__ void k_cvt_transpose_f16(const float* __restrict__ W, _Float16* __restrict__ BT,
                                    int K, int Ncols) {
  int i = blockIdx.x * blockDim.x + threadIdx.x;
  if (i >= K * Ncols) return;
  int k = i / Ncols, c = i % Ncols;
  BT[(size_t)c * K + k] = (_Float16)W[i];
}

// elu(x + bias[c]) -> f16 (feeds next WMMA GEMM)
__global__ void k_elu_bias_f16(const float* __restrict__ in, const float* __restrict__ bias,
                               _Float16* __restrict__ out, int total, int F) {
  int i = blockIdx.x * blockDim.x + threadIdx.x;
  if (i >= total) return;
  float v = in[i] + bias[i % F];
  v = v > 0.0f ? v : (__expf(v) - 1.0f);
  out[i] = (_Float16)v;
}

// ---------- WMMA GEMM: C[M,NCOLS] = A[M,K](f16,row) * B(f16, pre-transposed BT[NCOLS,K]) ----
// One wave computes one 16x16 tile; waves within a block cover column tiles.
// A frag (16-bit 16x32): lane r=lane&15 -> row M=r; lanes<16 K:{0..7,16..23},
// lanes>=16 K:{8..15,24..31}. B frag: col=lane&15, same K split as A.
// With BT, both fragments are contiguous 16B vector loads.
template <int K, int NCOLS>
__global__ void k_wmma_gemm(const _Float16* __restrict__ A, const _Float16* __restrict__ BT,
                            float* __restrict__ C, int M) {
  const int lane = threadIdx.x & 31;
  const int wave = threadIdx.x >> 5;
  const int row0 = blockIdx.x << 4;
  const int col0 = wave << 4;
  const int r    = lane & 15;      // A row within tile, and B column within tile
  const int hi   = lane >> 4;      // 0 or 1
  int ra = row0 + r; if (ra >= M) ra = M - 1;   // clamp A reads

  const _Float16* Ap = A  + (size_t)ra * K;
  const _Float16* Bp = BT + (size_t)(col0 + r) * K;

  v8f acc = {};
#pragma unroll
  for (int k0 = 0; k0 < K; k0 += 32) {
    v8h alo = *(const v8h*)(Ap + k0 + hi * 8);
    v8h ahi = *(const v8h*)(Ap + k0 + hi * 8 + 16);
    v8h blo = *(const v8h*)(Bp + k0 + hi * 8);
    v8h bhi = *(const v8h*)(Bp + k0 + hi * 8 + 16);
    v16h a, b;
#pragma unroll
    for (int i = 0; i < 8; ++i) {
      a[i] = alo[i]; a[i + 8] = ahi[i];
      b[i] = blo[i]; b[i + 8] = bhi[i];
    }
    acc = __builtin_amdgcn_wmma_f32_16x16x32_f16(false, a, false, b, (short)0, acc,
                                                 false, false);
  }
  const int rw = row0 + hi * 8;
  float* Cp = C + (size_t)rw * NCOLS + col0 + r;
  if (row0 + 16 <= M) {            // wave-uniform fast path: branchless stores
#pragma unroll
    for (int i = 0; i < 8; ++i) Cp[(size_t)i * NCOLS] = acc[i];
  } else {                         // generic tail tile
#pragma unroll
    for (int i = 0; i < 8; ++i)
      if (rw + i < M) Cp[(size_t)i * NCOLS] = acc[i];
  }
}

// ---------- per-node attention coefficients: as[n,h]=dot(h[n,h,:],a_src[h,:]) ----------
template <int H, int C>
__global__ void k_alpha(const float* __restrict__ h, const float* __restrict__ a_src,
                        const float* __restrict__ a_dst, float* __restrict__ as_,
                        float* __restrict__ ad_, int total /*N*H*/) {
  int i = blockIdx.x * blockDim.x + threadIdx.x;
  if (i >= total) return;
  int n = i / H, hh = i % H;
  const float* hp = h + (size_t)n * (H * C) + hh * C;
  float s = 0.0f, d = 0.0f;
#pragma unroll
  for (int c = 0; c < C; ++c) { s += hp[c] * a_src[hh * C + c]; d += hp[c] * a_dst[hh * C + c]; }
  as_[i] = s; ad_[i] = d;
}

// ---------- edge passes (segment softmax) ----------
template <int H>
__global__ void k_edge_max(const int* __restrict__ src, const int* __restrict__ dst,
                           const float* __restrict__ as_, const float* __restrict__ ad_,
                           unsigned* __restrict__ m, int E, int Etot) {
  int e = blockIdx.x * blockDim.x + threadIdx.x;
  if (e >= Etot) return;
  int s, d;
  if (e < E) { s = src[e]; d = dst[e]; } else { s = d = e - E; }
#pragma unroll
  for (int h = 0; h < H; ++h) {
    float v = lrelu(as_[s * H + h] + ad_[d * H + h]);
    atomicMax(&m[d * H + h], fkey(v));
  }
}

template <int H>
__global__ void k_edge_sum(const int* __restrict__ src, const int* __restrict__ dst,
                           const float* __restrict__ as_, const float* __restrict__ ad_,
                           const unsigned* __restrict__ m, float* __restrict__ den,
                           int E, int Etot) {
  int e = blockIdx.x * blockDim.x + threadIdx.x;
  if (e >= Etot) return;
  int s, d;
  if (e < E) { s = src[e]; d = dst[e]; } else { s = d = e - E; }
#pragma unroll
  for (int h = 0; h < H; ++h) {
    float v = lrelu(as_[s * H + h] + ad_[d * H + h]);
    atomicAdd(&den[d * H + h], __expf(v - fdec(m[d * H + h])));
  }
}

// one thread per (edge, head): out[d,h,:] += alpha * hfeat[s,h,:]
template <int H, int C>
__global__ void k_edge_agg(const int* __restrict__ src, const int* __restrict__ dst,
                           const float* __restrict__ as_, const float* __restrict__ ad_,
                           const unsigned* __restrict__ m, const float* __restrict__ den,
                           const float* __restrict__ hfeat, float* __restrict__ out,
                           int E, int Etot) {
  int t = blockIdx.x * blockDim.x + threadIdx.x;
  if (t >= Etot * H) return;
  int e = t / H, h = t % H;
  int s, d;
  if (e < E) { s = src[e]; d = dst[e]; } else { s = d = e - E; }
  float v = lrelu(as_[s * H + h] + ad_[d * H + h]);
  float w = __expf(v - fdec(m[d * H + h])) / (den[d * H + h] + 1e-16f);
  const float4* hp = (const float4*)(hfeat + (size_t)s * (H * C) + h * C);
  float* op = out + (size_t)d * (H * C) + h * C;
#pragma unroll
  for (int q = 0; q < C / 4; ++q) {
    float4 vv = hp[q];
    atomicAdd(&op[q * 4 + 0], w * vv.x);
    atomicAdd(&op[q * 4 + 1], w * vv.y);
    atomicAdd(&op[q * 4 + 2], w * vv.z);
    atomicAdd(&op[q * 4 + 3], w * vv.w);
  }
}

// ---------- classifier + softmax + argmax, writes all four outputs ----------
__global__ void k_classifier(const float* __restrict__ emb_in, const float* __restrict__ b2,
                             const float* __restrict__ Wc, const float* __restrict__ bc,
                             float* __restrict__ out, int N) {
  int n = blockIdx.x * blockDim.x + threadIdx.x;
  if (n >= N) return;
  float* logits = out;
  float* embO   = out + (size_t)N * OUT_F;
  float* soft   = out + (size_t)N * (OUT_F + HID);
  float* hard   = out + (size_t)N * (OUT_F + HID + OUT_F);
  float e16[HID];
#pragma unroll
  for (int c = 0; c < HID; ++c) {
    e16[c] = emb_in[(size_t)n * HID + c] + b2[c];
    embO[(size_t)n * HID + c] = e16[c];
  }
  float lg[OUT_F];
#pragma unroll
  for (int j = 0; j < OUT_F; ++j) {
    float a = bc[j];
#pragma unroll
    for (int c = 0; c < HID; ++c) a += e16[c] * Wc[c * OUT_F + j];
    lg[j] = a;
    logits[(size_t)n * OUT_F + j] = a;
  }
  float mx = lg[0]; int bi = 0;
#pragma unroll
  for (int j = 1; j < OUT_F; ++j) if (lg[j] > mx) { mx = lg[j]; bi = j; }
  float sum = 0.0f;
#pragma unroll
  for (int j = 0; j < OUT_F; ++j) { lg[j] = __expf(lg[j] - mx); sum += lg[j]; }
  float inv = 1.0f / sum;
#pragma unroll
  for (int j = 0; j < OUT_F; ++j) soft[(size_t)n * OUT_F + j] = lg[j] * inv;
  hard[n] = (float)bi;
}

// ---------------------------------------------------------------------------
extern "C" void kernel_launch(void* const* d_in, const int* in_sizes, int n_in,
                              void* d_out, int out_size, void* d_ws, size_t ws_size,
                              hipStream_t stream) {
  const float* x     = (const float*)d_in[0];
  const int*   eidx  = (const int*)d_in[1];      // [2,E] flattened: src then dst
  const float* W1    = (const float*)d_in[2];
  const float* a1s   = (const float*)d_in[3];
  const float* a1d   = (const float*)d_in[4];
  const float* b1    = (const float*)d_in[5];
  const float* W2    = (const float*)d_in[6];
  const float* a2s   = (const float*)d_in[7];
  const float* a2d   = (const float*)d_in[8];
  const float* b2    = (const float*)d_in[9];
  const float* Wc    = (const float*)d_in[10];
  const float* bc    = (const float*)d_in[11];
  float* out = (float*)d_out;

  const int N    = in_sizes[0] / IN_F;
  const int E    = in_sizes[1] / 2;
  const int Etot = E + N;                        // self-loops appended
  const int* src = eidx;
  const int* dst = eidx + E;

  // ---- workspace carve-out (256B aligned) ----
  size_t off = 0;
  auto take = [&](size_t bytes) -> void* {
    void* p = (char*)d_ws + off;
    off += (bytes + 255) & ~(size_t)255;
    return p;
  };
  _Float16* xh   = (_Float16*)take((size_t)N * IN_F * 2);   // f16 x, later f16 elu(out1 + b1)
  _Float16* WT   = (_Float16*)take((size_t)IN_F * IN_F * 2);// f16 W1^T, later W2^T
  float*    h1   = (float*)take((size_t)N * IN_F * 4);      // x@W1  [N,128]
  float*    out1 = (float*)take((size_t)N * IN_F * 4);      // layer1 aggregate
  float*    as1  = (float*)take((size_t)N * NHEAD * 4);
  float*    ad1  = (float*)take((size_t)N * NHEAD * 4);
  unsigned* m1   = (unsigned*)take((size_t)N * NHEAD * 4);
  float*    den1 = (float*)take((size_t)N * NHEAD * 4);
  float*    t2   = (float*)take((size_t)N * HID * 4);       // h2@W2 [N,16]
  float*    out2 = (float*)take((size_t)N * HID * 4);       // layer2 aggregate = emb
  float*    as2  = (float*)take((size_t)N * 4);
  float*    ad2  = (float*)take((size_t)N * 4);
  unsigned* m2   = (unsigned*)take((size_t)N * 4);
  float*    den2 = (float*)take((size_t)N * 4);

  auto cdiv = [](int a, int b) { return (a + b - 1) / b; };
  const int BLK = 256;

  // ---- zero accumulators / max keys (key 0 == most-negative ordering) ----
  k_zero_u32<<<cdiv(N * NHEAD, BLK), BLK, 0, stream>>>(m1, N * NHEAD);
  k_zero_u32<<<cdiv(N * NHEAD, BLK), BLK, 0, stream>>>((unsigned*)den1, N * NHEAD);
  k_zero_u32<<<cdiv(N * IN_F, BLK), BLK, 0, stream>>>((unsigned*)out1, N * IN_F);
  k_zero_u32<<<cdiv(N, BLK), BLK, 0, stream>>>(m2, N);
  k_zero_u32<<<cdiv(N, BLK), BLK, 0, stream>>>((unsigned*)den2, N);
  k_zero_u32<<<cdiv(N * HID, BLK), BLK, 0, stream>>>((unsigned*)out2, N * HID);

  // ---- layer 1: h1 = x @ W1 via WMMA (W1 pre-transposed to f16) ----
  k_f32_to_f16<<<cdiv(N * IN_F, BLK), BLK, 0, stream>>>(x, xh, N * IN_F);
  k_cvt_transpose_f16<<<cdiv(IN_F * IN_F, BLK), BLK, 0, stream>>>(W1, WT, IN_F, IN_F);
  k_wmma_gemm<IN_F, IN_F><<<cdiv(N, 16), (IN_F / 16) * 32, 0, stream>>>(xh, WT, h1, N);

  k_alpha<NHEAD, HID><<<cdiv(N * NHEAD, BLK), BLK, 0, stream>>>(h1, a1s, a1d, as1, ad1,
                                                                N * NHEAD);
  k_edge_max<NHEAD><<<cdiv(Etot, BLK), BLK, 0, stream>>>(src, dst, as1, ad1, m1, E, Etot);
  k_edge_sum<NHEAD><<<cdiv(Etot, BLK), BLK, 0, stream>>>(src, dst, as1, ad1, m1, den1, E, Etot);
  k_edge_agg<NHEAD, HID><<<cdiv(Etot * NHEAD, BLK), BLK, 0, stream>>>(src, dst, as1, ad1, m1,
                                                                      den1, h1, out1, E, Etot);

  // ---- elu(out1 + b1) -> f16 (reuse xh), then t2 = h2 @ W2 via WMMA ----
  k_elu_bias_f16<<<cdiv(N * IN_F, BLK), BLK, 0, stream>>>(out1, b1, xh, N * IN_F, IN_F);
  k_cvt_transpose_f16<<<cdiv(IN_F * HID, BLK), BLK, 0, stream>>>(W2, WT, IN_F, HID);
  k_wmma_gemm<IN_F, HID><<<cdiv(N, 16), (HID / 16) * 32, 0, stream>>>(xh, WT, t2, N);

  k_alpha<1, HID><<<cdiv(N, BLK), BLK, 0, stream>>>(t2, a2s, a2d, as2, ad2, N);
  k_edge_max<1><<<cdiv(Etot, BLK), BLK, 0, stream>>>(src, dst, as2, ad2, m2, E, Etot);
  k_edge_sum<1><<<cdiv(Etot, BLK), BLK, 0, stream>>>(src, dst, as2, ad2, m2, den2, E, Etot);
  k_edge_agg<1, HID><<<cdiv(Etot, BLK), BLK, 0, stream>>>(src, dst, as2, ad2, m2, den2, t2,
                                                          out2, E, Etot);

  // ---- classifier + softmax + argmax; writes logits/emb/soft/hard ----
  k_classifier<<<cdiv(N, BLK), BLK, 0, stream>>>(out2, b2, Wc, bc, out, N);
  (void)n_in; (void)out_size; (void)ws_size;
}
